// Graph2SeqGenerator_7962869366792
// MI455X (gfx1250) — compile-verified, matrix operand order checked by reading, not run on hardware
//
#include <hip/hip_runtime.h>

#define NN    10000
#define NE    160000
#define NE2   170000
#define VV    50257
#define VPAD  50272
#define EDIM  256
#define HIDN  128
#define TLEN  64

typedef __attribute__((ext_vector_type(16))) __bf16 v16bf;
typedef __attribute__((ext_vector_type(8)))  __bf16 v8bf;
typedef __attribute__((ext_vector_type(8)))  float  v8f;

// ---------- helpers ----------
__device__ __forceinline__ unsigned ford(float f) {
  unsigned u = __float_as_uint(f);
  return (u & 0x80000000u) ? ~u : (u | 0x80000000u);   // order-preserving map
}
__device__ __forceinline__ float ordf(unsigned u) {
  return (u & 0x80000000u) ? __uint_as_float(u ^ 0x80000000u) : __uint_as_float(~u);
}
__device__ __forceinline__ void edge_sd(const int* ei, int e, int& s, int& d) {
  if (e < NE) { s = ei[e]; d = ei[NE + e]; } else { s = e - NE; d = s; }
}

// ---------- init / packing ----------
__global__ void zero_u32(unsigned* p, size_t n) {
  size_t i = (size_t)blockIdx.x * blockDim.x + threadIdx.x;
  size_t st = (size_t)gridDim.x * blockDim.x;
  for (; i < n; i += st) p[i] = 0u;
}

// W[K][N] f32 (row-major) -> Wt[N][K] bf16
__global__ void pack_wT_bf16(const float* __restrict__ W, __bf16* __restrict__ Wt,
                             int K, int N) {
  int i = blockIdx.x * blockDim.x + threadIdx.x;
  if (i >= K * N) return;
  int k = i / N, n = i % N;
  Wt[(size_t)n * K + k] = (__bf16)W[i];
}

// Wout[V][128] f32 -> bf16 padded to VPAD rows (already [N][K] layout)
__global__ void pack_wout_bf16(const float* __restrict__ W, __bf16* __restrict__ Wt) {
  size_t i = (size_t)blockIdx.x * blockDim.x + threadIdx.x;
  if (i >= (size_t)VPAD * HIDN) return;
  Wt[i] = (i < (size_t)VV * HIDN) ? (__bf16)W[i] : (__bf16)0.0f;
}

// feats_bf[n][c] = bf16(embedding[x_nodes[n]][c])
__global__ void gather_embed_bf16(const int* __restrict__ xn,
                                  const float* __restrict__ emb,
                                  __bf16* __restrict__ out) {
  int i = blockIdx.x * blockDim.x + threadIdx.x;
  if (i >= NN * EDIM) return;
  int n = i >> 8, c = i & 255;
  out[i] = (__bf16)emb[(size_t)xn[n] * EDIM + c];
}

// ---------- WMMA GEMM: C[M][ldc] = A[M][K](bf16) * Bt[N][K](bf16)^T (+bias) ----------
__global__ void wmma_gemm_bf16(const __bf16* __restrict__ A,
                               const __bf16* __restrict__ Bt,
                               float* __restrict__ C,
                               const float* __restrict__ bias,
                               int M, int N, int K, int ldc) {
  int wave = (blockIdx.x * blockDim.x + threadIdx.x) >> 5;
  int lane = threadIdx.x & 31;
  int tilesN = (N + 15) >> 4;
  int total  = (M >> 4) * tilesN;
  if (wave >= total) return;                 // whole-wave exit; EXEC stays all-1 for WMMA
  int m0 = (wave / tilesN) << 4;
  int n0 = (wave % tilesN) << 4;
  int r   = lane & 15;
  int hi8 = (lane >> 4) << 3;                // 0 or 8 (K-half select per ISA A/B layout)
  const __bf16* arow = A  + (size_t)(m0 + r) * K + hi8;
  const __bf16* brow = Bt + (size_t)(n0 + r) * K + hi8;
  v8f acc = {};
  for (int k0 = 0; k0 < K; k0 += 32) {
    v8bf al = *(const v8bf*)(arow + k0);
    v8bf ah = *(const v8bf*)(arow + k0 + 16);
    v8bf bl = *(const v8bf*)(brow + k0);
    v8bf bh = *(const v8bf*)(brow + k0 + 16);
    v16bf a, b;
#pragma unroll
    for (int i = 0; i < 8; ++i) { a[i] = al[i]; a[i + 8] = ah[i]; b[i] = bl[i]; b[i + 8] = bh[i]; }
    acc = __builtin_amdgcn_wmma_f32_16x16x32_bf16(false, a, false, b, (short)0, acc, false, false);
  }
  int col = n0 + (lane & 15);
  int hi  = lane >> 4;
  if (col < N) {
    float bv = bias ? bias[col] : 0.0f;
#pragma unroll
    for (int g = 0; g < 8; ++g) {
      int row = m0 + g + (hi << 3);
      C[(size_t)row * ldc + col] = acc[g] + bv;
    }
  }
}

// ---------- GATv2 edge kernels ----------
__global__ void edge_logits(const int* __restrict__ ei,
                            const float* __restrict__ xl, const float* __restrict__ xr,
                            const float* __restrict__ att,
                            float* __restrict__ lg, unsigned* __restrict__ smax,
                            int heads, int ch) {
  int tid = blockIdx.x * blockDim.x + threadIdx.x;
  if (tid >= NE2 * heads) return;
  int e = tid / heads, hh = tid % heads;
  int s, d; edge_sd(ei, e, s, d);
  int stride = heads * ch;
  const float4* pl = (const float4*)(xl + (size_t)s * stride + hh * ch);
  const float4* pr = (const float4*)(xr + (size_t)d * stride + hh * ch);
  const float4* pa = (const float4*)(att + hh * ch);
  float sum = 0.0f;
  for (int q = 0; q < (ch >> 2); ++q) {
    float4 a = pl[q], b = pr[q], w = pa[q];
    float v0 = a.x + b.x; v0 = v0 > 0.f ? v0 : 0.2f * v0;
    float v1 = a.y + b.y; v1 = v1 > 0.f ? v1 : 0.2f * v1;
    float v2 = a.z + b.z; v2 = v2 > 0.f ? v2 : 0.2f * v2;
    float v3 = a.w + b.w; v3 = v3 > 0.f ? v3 : 0.2f * v3;
    sum += w.x * v0 + w.y * v1 + w.z * v2 + w.w * v3;
  }
  lg[tid] = sum;
  atomicMax(&smax[d * heads + hh], ford(sum));
}

__global__ void edge_exp(const int* __restrict__ ei,
                         float* __restrict__ lg,
                         const unsigned* __restrict__ smax,
                         float* __restrict__ ssum, int heads) {
  int tid = blockIdx.x * blockDim.x + threadIdx.x;
  if (tid >= NE2 * heads) return;
  int e = tid / heads, hh = tid % heads;
  int s, d; edge_sd(ei, e, s, d);
  float m = ordf(smax[d * heads + hh]);
  float a = __expf(lg[tid] - m);
  lg[tid] = a;                               // reuse buffer for numerator
  atomicAdd(&ssum[d * heads + hh], a);
}

__global__ void edge_aggregate(const int* __restrict__ ei,
                               const float* __restrict__ xl,
                               const float* __restrict__ a,
                               const float* __restrict__ ssum,
                               float* __restrict__ out, int heads, int ch) {
  int quads = ch >> 2;
  int per = heads * quads;
  long long tid = (long long)blockIdx.x * blockDim.x + threadIdx.x;
  if (tid >= (long long)NE2 * per) return;
  int e = (int)(tid / per);
  int rem = (int)(tid % per);
  int hh = rem / quads, q = rem % quads;
  int s, d; edge_sd(ei, e, s, d);
  float w = a[e * heads + hh] / ssum[d * heads + hh];
  int stride = heads * ch;
  float4 v = *(const float4*)(xl + (size_t)s * stride + hh * ch + q * 4);
  float* o = out + (size_t)d * stride + hh * ch + q * 4;
  atomicAdd(o + 0, v.x * w); atomicAdd(o + 1, v.y * w);
  atomicAdd(o + 2, v.z * w); atomicAdd(o + 3, v.w * w);
}

// ---------- post-aggregation ----------
__global__ void finalize1(const float* __restrict__ h1pre, const float* __restrict__ b1,
                          __bf16* __restrict__ h1bf) {
  int i = blockIdx.x * blockDim.x + threadIdx.x;
  if (i >= NN * 512) return;
  float v = h1pre[i] + b1[i & 511];
  v = v > 0.f ? v : (__expf(v) - 1.0f);      // ELU
  h1bf[i] = (__bf16)v;
}

__global__ void finalize2_ctx(const float* __restrict__ encpre, const float* __restrict__ b2,
                              float* __restrict__ ctx) {
  int i = blockIdx.x * blockDim.x + threadIdx.x;
  if (i >= NN * HIDN) return;
  float v = encpre[i] + b2[i & 127];
  atomicAdd(&ctx[i & 127], v * (1.0f / (float)NN));   // mean over nodes
}

// ---------- GRU decoder (sequential, one block) ----------
__global__ void gru_kernel(const float* __restrict__ emb, const int* __restrict__ tseq,
                           const float* __restrict__ ctx,
                           const float* __restrict__ Wih, const float* __restrict__ Whh,
                           const float* __restrict__ bih, const float* __restrict__ bhh,
                           __bf16* __restrict__ Hbf) {
  __shared__ float x[384], h[128], hn[128], gi[384], gh[384];
  int j = threadIdx.x;                       // 0..383
  if (j < 128) h[j] = 0.0f;
  if (j >= 256) x[j] = ctx[j - 256];         // x = [embed(256), context(128)]
  __syncthreads();
  for (int t = 0; t < TLEN; ++t) {
    int tok = (t == 0) ? 0 : tseq[t - 1];    // teacher forcing, SOS=0
    if (j < 256) x[j] = emb[(size_t)tok * EDIM + j];
    __syncthreads();
    float s = bih[j];
    const float* wr = Wih + (size_t)j * 384;
    for (int k = 0; k < 384; ++k) s += wr[k] * x[k];
    gi[j] = s;
    float s2 = bhh[j];
    const float* wr2 = Whh + (size_t)j * 128;
    for (int k = 0; k < 128; ++k) s2 += wr2[k] * h[k];
    gh[j] = s2;
    __syncthreads();
    if (j < 128) {
      float r = 1.0f / (1.0f + __expf(-(gi[j] + gh[j])));
      float z = 1.0f / (1.0f + __expf(-(gi[128 + j] + gh[128 + j])));
      float c = tanhf(gi[256 + j] + r * gh[256 + j]);
      float v = (1.0f - z) * c + z * h[j];
      hn[j] = v;
      Hbf[t * HIDN + j] = (__bf16)v;
    }
    __syncthreads();
    if (j < 128) h[j] = hn[j];
    __syncthreads();
  }
}

// ---------- host ----------
extern "C" void kernel_launch(void* const* d_in, const int* in_sizes, int n_in,
                              void* d_out, int out_size, void* d_ws, size_t ws_size,
                              hipStream_t stream) {
  const int*   x_nodes = (const int*)d_in[0];
  const int*   ei      = (const int*)d_in[1];
  const int*   tseq    = (const int*)d_in[2];
  const float* emb     = (const float*)d_in[3];
  const float* W1l     = (const float*)d_in[4];
  const float* W1r     = (const float*)d_in[5];
  const float* att1    = (const float*)d_in[6];
  const float* b1      = (const float*)d_in[7];
  const float* W2l     = (const float*)d_in[8];
  const float* W2r     = (const float*)d_in[9];
  const float* att2    = (const float*)d_in[10];
  const float* b2      = (const float*)d_in[11];
  const float* Wih     = (const float*)d_in[12];
  const float* Whh     = (const float*)d_in[13];
  const float* bih     = (const float*)d_in[14];
  const float* bhh     = (const float*)d_in[15];
  const float* Wout    = (const float*)d_in[16];
  const float* bout    = (const float*)d_in[17];
  float* out = (float*)d_out;

  char* ws = (char*)d_ws;
  size_t off = 0;
  auto alloc = [&](size_t b) -> char* {
    char* p = ws + off; off = (off + b + 255) & ~(size_t)255; return p;
  };
  // --- zero zone (cleared every launch) ---
  float*    h1pre  = (float*)   alloc((size_t)NN * 512 * 4);
  float*    encpre = (float*)   alloc((size_t)NN * 128 * 4);
  unsigned* smax1  = (unsigned*)alloc((size_t)NN * 4 * 4);
  float*    ssum1  = (float*)   alloc((size_t)NN * 4 * 4);
  unsigned* smax2  = (unsigned*)alloc((size_t)NN * 4);
  float*    ssum2  = (float*)   alloc((size_t)NN * 4);
  float*    ctx    = (float*)   alloc(128 * 4);
  size_t zero_words = off / 4;
  // --- persistent-per-launch scratch ---
  __bf16* featsbf = (__bf16*)alloc((size_t)NN * EDIM * 2);
  __bf16* w1lt    = (__bf16*)alloc((size_t)512 * 256 * 2);
  __bf16* w1rt    = (__bf16*)alloc((size_t)512 * 256 * 2);
  __bf16* w2lt    = (__bf16*)alloc((size_t)128 * 512 * 2);
  __bf16* w2rt    = (__bf16*)alloc((size_t)128 * 512 * 2);
  __bf16* woutbf  = (__bf16*)alloc((size_t)VPAD * HIDN * 2);
  float*  xl1     = (float*) alloc((size_t)NN * 512 * 4);
  float*  xr1     = (float*) alloc((size_t)NN * 512 * 4);
  float*  a1      = (float*) alloc((size_t)NE2 * 4 * 4);
  __bf16* h1bf    = (__bf16*)alloc((size_t)NN * 512 * 2);
  float*  xl2     = (float*) alloc((size_t)NN * 128 * 4);
  float*  xr2     = (float*) alloc((size_t)NN * 128 * 4);
  float*  a2      = (float*) alloc((size_t)NE2 * 4);
  __bf16* Hbf     = (__bf16*)alloc((size_t)TLEN * HIDN * 2);

  auto cdiv = [](long long a, long long b) { return (int)((a + b - 1) / b); };
  auto gemm = [&](const __bf16* A, const __bf16* Bt, float* C, const float* bias,
                  int M, int N, int K, int ldc) {
    long long waves = (long long)(M >> 4) * ((N + 15) >> 4);
    wmma_gemm_bf16<<<cdiv(waves * 32, 256), 256, 0, stream>>>(A, Bt, C, bias, M, N, K, ldc);
  };

  // 0) clear accumulators / segment state / ctx
  zero_u32<<<2048, 256, 0, stream>>>((unsigned*)ws, zero_words);
  // 1) pack weights (transposed bf16) + padded bf16 Wout
  pack_wT_bf16<<<cdiv(256 * 512, 256), 256, 0, stream>>>(W1l, w1lt, 256, 512);
  pack_wT_bf16<<<cdiv(256 * 512, 256), 256, 0, stream>>>(W1r, w1rt, 256, 512);
  pack_wT_bf16<<<cdiv(512 * 128, 256), 256, 0, stream>>>(W2l, w2lt, 512, 128);
  pack_wT_bf16<<<cdiv(512 * 128, 256), 256, 0, stream>>>(W2r, w2rt, 512, 128);
  pack_wout_bf16<<<cdiv((long long)VPAD * HIDN, 256), 256, 0, stream>>>(Wout, woutbf);
  // 2) embedding gather (bf16 for WMMA A)
  gather_embed_bf16<<<cdiv(NN * EDIM, 256), 256, 0, stream>>>(x_nodes, emb, featsbf);
  // 3) GATv2 layer 1 (heads=4, ch=128)
  gemm(featsbf, w1lt, xl1, nullptr, NN, 512, 256, 512);
  gemm(featsbf, w1rt, xr1, nullptr, NN, 512, 256, 512);
  edge_logits<<<cdiv((long long)NE2 * 4, 256), 256, 0, stream>>>(ei, xl1, xr1, att1, a1, smax1, 4, 128);
  edge_exp<<<cdiv((long long)NE2 * 4, 256), 256, 0, stream>>>(ei, a1, smax1, ssum1, 4);
  edge_aggregate<<<cdiv((long long)NE2 * 4 * 32, 256), 256, 0, stream>>>(ei, xl1, a1, ssum1, h1pre, 4, 128);
  finalize1<<<cdiv(NN * 512, 256), 256, 0, stream>>>(h1pre, b1, h1bf);
  // 4) GATv2 layer 2 (heads=1, ch=128)
  gemm(h1bf, w2lt, xl2, nullptr, NN, 128, 512, 128);
  gemm(h1bf, w2rt, xr2, nullptr, NN, 128, 512, 128);
  edge_logits<<<cdiv((long long)NE2, 256), 256, 0, stream>>>(ei, xl2, xr2, att2, a2, smax2, 1, 128);
  edge_exp<<<cdiv((long long)NE2, 256), 256, 0, stream>>>(ei, a2, smax2, ssum2, 1);
  edge_aggregate<<<cdiv((long long)NE2 * 32, 256), 256, 0, stream>>>(ei, xl2, a2, ssum2, encpre, 1, 128);
  finalize2_ctx<<<cdiv(NN * HIDN, 256), 256, 0, stream>>>(encpre, b2, ctx);
  // 5) GRU decode (sequential) -> Hbf [64][128]
  gru_kernel<<<1, 384, 0, stream>>>(emb, tseq, ctx, Wih, Whh, bih, bhh, Hbf);
  // 6) logits = H @ Wout^T + bout  -> d_out [64][50257]
  gemm(Hbf, woutbf, out, bout, TLEN, VV, HIDN, VV);
}
